// GFeedforward_7842610283358
// MI455X (gfx1250) — matrix operand order repaired and checked by reading, not compiled, not run on hardware
//
#include <hip/hip_runtime.h>
#include <hip/hip_bf16.h>
#include <math.h>

// ---------------------------------------------------------------------------
// Fused embedding-gather -> [B,992] x -> Linear(992,256) -> tanh(x^3)
//       -> Linear(256,50) for MI455X (gfx1250, wave32, WMMA bf16,
//       async global->LDS weight staging, double-buffered).
// ---------------------------------------------------------------------------

typedef __attribute__((ext_vector_type(16))) __bf16 v16bf;
typedef __attribute__((ext_vector_type(8)))  float  v8f;

#define NUM_FEAT   10
#define WORD_DIM   64
#define POS_DIM    32
#define DIS_DIM    32
#define IN_SIZE    992     // 10*64 + 10*32 + 32
#define HIDDEN     256
#define OUT_SIZE   50
#define OUT_PAD    64      // pad 50 -> 64 for clean 16-wide WMMA tiles

#define WAVE          32
#define WAVES_PER_BLK 4
#define THREADS       (WAVE * WAVES_PER_BLK)           // 128
#define ROWS_PER_WAVE 16
#define ROWS_PER_BLK  (ROWS_PER_WAVE * WAVES_PER_BLK)  // 64

#define KSTEP      32                 // K per WMMA
#define NKSTEPS    (IN_SIZE / KSTEP)  // 31
#define NCHUNK     64                 // hidden cols per chunk
#define SLAB_ELTS  (NCHUNK * KSTEP)   // 2048 bf16 = 4 KB per slab

__device__ __forceinline__ v8f zero8() {
    v8f z = {0.f, 0.f, 0.f, 0.f, 0.f, 0.f, 0.f, 0.f};
    return z;
}

union Frag {
    v16bf v;
    uint4 u[2];
};

// ---- CDNA5 async global->LDS copy (ASYNCcnt-tracked, no VGPR round trip) --
__device__ __forceinline__ void async_copy_b128(const void* gptr, void* lptr) {
    // low 32 bits of a generic LDS-aperture address are the LDS byte offset
    const unsigned lds_off = (unsigned)(unsigned long long)lptr;
    asm volatile("global_load_async_to_lds_b128 %0, %1, off"
                 :: "v"(lds_off), "v"(gptr) : "memory");
}

__device__ __forceinline__ void wait_async0() {
    asm volatile("s_wait_asynccnt 0x0" ::: "memory");
}

// A-matrix fragment (16x32 bf16):
//   lanes 0-15  : M=lane,    elems j -> K=kbase+j
//   lanes 16-31 : M=lane-16, elems j -> K=kbase+16+j
__device__ __forceinline__ v16bf load_a_frag(const __bf16* __restrict__ base,
                                             int strideElts, int kbase, int lane) {
    const int m  = lane & 15;
    const int kh = (lane >> 4) * 16;
    const __bf16* p = base + m * strideElts + kbase + kh;
    Frag f;
    f.u[0] = *(const uint4*)(p);
    f.u[1] = *(const uint4*)(p + 8);
    return f.v;
}

// B fragment (32x16) from LDS slab laid out [64 n][32 k]:
//   lanes 0-15: N=t*16+lane, elems j -> K=j; lanes 16-31: same N, K=16+j
__device__ __forceinline__ v16bf load_b_frag_lds(const __bf16* slab, int t, int lane) {
    const int n  = t * 16 + (lane & 15);
    const int kh = (lane >> 4) * 16;
    const __bf16* p = slab + n * KSTEP + kh;
    Frag f;
    f.u[0] = *(const uint4*)(p);
    f.u[1] = *(const uint4*)(p + 8);
    return f.v;
}

// B fragment (32x16) straight from N-major global weights (layer 2, tiny)
__device__ __forceinline__ v16bf load_b_frag(const __bf16* __restrict__ Wt,
                                             int strideElts, int nbase, int kbase,
                                             int lane) {
    const int n  = nbase + (lane & 15);
    const int kh = (lane >> 4) * 16;
    const __bf16* p = Wt + n * strideElts + kbase + kh;
    Frag f;
    f.u[0] = *(const uint4*)(p);
    f.u[1] = *(const uint4*)(p + 8);
    return f.v;
}

// ---------------------------------------------------------------------------
// Once-per-launch weight prep: transpose to N-major and convert fp32 -> bf16.
// ---------------------------------------------------------------------------
__global__ void convert_weights(const float* __restrict__ w1,
                                const float* __restrict__ w2,
                                __bf16* __restrict__ w1t,
                                __bf16* __restrict__ w2t) {
    const int n1 = HIDDEN * IN_SIZE;
    const int n2 = OUT_PAD * HIDDEN;
    for (int idx = blockIdx.x * blockDim.x + threadIdx.x; idx < n1 + n2;
         idx += gridDim.x * blockDim.x) {
        if (idx < n1) {
            int n = idx / IN_SIZE;
            int k = idx - n * IN_SIZE;
            w1t[idx] = (__bf16)w1[k * HIDDEN + n];
        } else {
            int j = idx - n1;
            int n = j / HIDDEN;
            int k = j - n * HIDDEN;
            w2t[j] = (__bf16)((n < OUT_SIZE) ? w2[k * OUT_SIZE + n] : 0.0f);
        }
    }
}

// ---------------------------------------------------------------------------
// Main fused kernel.
//   static LDS : wslab[2][64][32] bf16 double buffer     (8192 B)
//   dynamic LDS: xs [64][992] bf16 (126976 B) + hs [4][16][256] bf16 (32768 B)
// ---------------------------------------------------------------------------
__global__ __launch_bounds__(THREADS) void gfeed_kernel(
    const int*   __restrict__ word_idx, const int* __restrict__ pos_idx,
    const int*   __restrict__ dis_idx,
    const float* __restrict__ word_emb, const float* __restrict__ pos_emb,
    const float* __restrict__ dis_emb,
    const __bf16* __restrict__ w1t, const float* __restrict__ b1,
    const __bf16* __restrict__ w2t, const float* __restrict__ b2,
    float* __restrict__ out)
{
    __shared__ __align__(16) __bf16 wslab[2][SLAB_ELTS];
    extern __shared__ __align__(16) __bf16 smem[];
    __bf16* xs = smem;                                   // [64][992]
    __bf16* hs = smem + ROWS_PER_BLK * IN_SIZE;          // [4][16][256]

    const int tid  = threadIdx.x;
    const int lane = tid & (WAVE - 1);
    const int wv   = tid >> 5;
    const int row0 = blockIdx.x * ROWS_PER_BLK;

    // warm layer-2 weights (32 KB) into cache: global_prefetch_b8
    __builtin_prefetch(((const char*)w2t) + (size_t)tid * 256, 0, 1);

    // ---- stage 1: cooperative gather + fp32->bf16 convert into LDS --------
    for (int r = 0; r < ROWS_PER_BLK; ++r) {
        const int grow = row0 + r;
        for (int it = 0; it < 8; ++it) {               // 8*128 = 1024 >= 992
            const int col = it * THREADS + tid;
            if (col < IN_SIZE) {
                float v;
                if (col < NUM_FEAT * WORD_DIM) {                    // [0,640)
                    const int f = col >> 6, d = col & 63;
                    v = word_emb[word_idx[grow * NUM_FEAT + f] * WORD_DIM + d];
                } else if (col < NUM_FEAT * (WORD_DIM + POS_DIM)) { // [640,960)
                    const int c = col - NUM_FEAT * WORD_DIM;
                    const int f = c >> 5, d = c & 31;
                    v = pos_emb[pos_idx[grow * NUM_FEAT + f] * POS_DIM + d];
                } else {                                            // [960,992)
                    v = dis_emb[dis_idx[grow] * DIS_DIM + (col - 960)];
                }
                xs[r * IN_SIZE + col] = (__bf16)v;
            }
        }
    }

    const __bf16* xw = xs + wv * ROWS_PER_WAVE * IN_SIZE;   // this wave's rows
    __bf16*       hw = hs + wv * ROWS_PER_WAVE * HIDDEN;

    // ---- async-stage w1 slab (chunk, kk): [64 n][32 k] -> wslab[buf] ------
    auto stage_slab = [&](int chunk, int kk, int buf) {
        #pragma unroll
        for (int i = 0; i < 2; ++i) {                 // 256 x 16B chunks total
            const int c  = i * THREADS + tid;         // 0..255
            const int nl = c >> 2;                    // 0..63
            const int ko = (c & 3) * 8;               // 0,8,16,24 elements
            const __bf16* src =
                w1t + (size_t)(chunk * NCHUNK + nl) * IN_SIZE + kk * KSTEP + ko;
            async_copy_b128(src, &wslab[buf][nl * KSTEP + ko]);
        }
    };

    stage_slab(0, 0, 0);                              // prologue fill
    int buf = 0;

    // ---- stage 2: layer 1, 16x256 per wave, N in 4 chunks of 64 -----------
    for (int chunk = 0; chunk < 4; ++chunk) {
        v8f acc[4];
        #pragma unroll
        for (int t = 0; t < 4; ++t) acc[t] = zero8();

        for (int kk = 0; kk < NKSTEPS; ++kk) {        // 31 K-steps
            wait_async0();                            // own async copies done
            __syncthreads();                          // slab visible block-wide
                                                      // (also: buf^1 free again)
            int nc = chunk, nk = kk + 1;              // issue next slab
            if (nk == NKSTEPS) { nk = 0; ++nc; }
            if (nc < 4) stage_slab(nc, nk, buf ^ 1);

            const v16bf a = load_a_frag(xw, IN_SIZE, kk * KSTEP, lane);
            #pragma unroll
            for (int t = 0; t < 4; ++t) {
                const v16bf b = load_b_frag_lds(wslab[buf], t, lane);
                acc[t] = __builtin_amdgcn_wmma_f32_16x16x32_bf16(
                    false, a, false, b, (short)0, acc[t], false, false);
            }
            buf ^= 1;
        }

        // bias + tanh(u^3), convert to bf16, stage into LDS for layer 2
        #pragma unroll
        for (int t = 0; t < 4; ++t) {
            const int n     = chunk * NCHUNK + t * 16 + (lane & 15);
            const float bia = b1[n];
            const int mbase = (lane >> 4) * 8;
            #pragma unroll
            for (int r = 0; r < 8; ++r) {
                const float u = acc[t][r] + bia;
                const float h = tanhf(u * u * u);
                hw[(mbase + r) * HIDDEN + n] = (__bf16)h;
            }
        }
    }

    // ---- stage 3: layer 2, 16x64 (50 valid) per wave, K=256 ---------------
    v8f oacc[4];
    #pragma unroll
    for (int t = 0; t < 4; ++t) oacc[t] = zero8();

    for (int kk = 0; kk < HIDDEN / KSTEP; ++kk) {     // 8 K-steps
        const int kbase = kk * KSTEP;
        const v16bf a = load_a_frag(hw, HIDDEN, kbase, lane);
        #pragma unroll
        for (int t = 0; t < 4; ++t) {
            const v16bf b = load_b_frag(w2t, HIDDEN, t * 16, kbase, lane);
            oacc[t] = __builtin_amdgcn_wmma_f32_16x16x32_bf16(
                false, a, false, b, (short)0, oacc[t], false, false);
        }
    }

    // ---- epilogue: bias + store the 50 valid columns ----------------------
    const int rowbase = row0 + wv * ROWS_PER_WAVE;
    #pragma unroll
    for (int t = 0; t < 4; ++t) {
        const int n = t * 16 + (lane & 15);
        if (n < OUT_SIZE) {
            const float bia = b2[n];
            const int mbase = (lane >> 4) * 8;
            #pragma unroll
            for (int r = 0; r < 8; ++r) {
                out[(size_t)(rowbase + mbase + r) * OUT_SIZE + n] =
                    oacc[t][r] + bia;
            }
        }
    }
}

// ---------------------------------------------------------------------------
extern "C" void kernel_launch(void* const* d_in, const int* in_sizes, int n_in,
                              void* d_out, int out_size, void* d_ws, size_t ws_size,
                              hipStream_t stream) {
    const int*   word_idx = (const int*)  d_in[0];
    const int*   pos_idx  = (const int*)  d_in[1];
    const int*   dis_idx  = (const int*)  d_in[2];
    const float* word_emb = (const float*)d_in[3];
    const float* pos_emb  = (const float*)d_in[4];
    const float* dis_emb  = (const float*)d_in[5];
    const float* w1       = (const float*)d_in[6];
    const float* b1       = (const float*)d_in[7];
    const float* w2       = (const float*)d_in[8];
    const float* b2       = (const float*)d_in[9];
    float*       out      = (float*)d_out;

    __bf16* w1t = (__bf16*)d_ws;                      // 256*992  bf16
    __bf16* w2t = w1t + HIDDEN * IN_SIZE;             //  64*256  bf16

    convert_weights<<<256, 256, 0, stream>>>(w1, w2, w1t, w2t);

    const int Bn     = in_sizes[2];                   // dis_idx has B elements
    const int blocks = Bn / ROWS_PER_BLK;             // 131072/64 = 2048
    const size_t lds =
        (size_t)(ROWS_PER_BLK * IN_SIZE + WAVES_PER_BLK * ROWS_PER_WAVE * HIDDEN)
        * sizeof(__bf16);                             // 159744 B dynamic

    gfeed_kernel<<<blocks, THREADS, lds, stream>>>(
        word_idx, pos_idx, dis_idx, word_emb, pos_emb, dis_emb,
        w1t, b1, w2t, b2, out);
}
